// MultiHeadAttention_23596550324456
// MI455X (gfx1250) — matrix operand order repaired and checked by reading, not compiled, not run on hardware
//
#include <hip/hip_runtime.h>
#include <hip/hip_bf16.h>

typedef __bf16 bf16_t;
typedef __attribute__((ext_vector_type(16))) __bf16 v16bf;
typedef __attribute__((ext_vector_type(8)))  __bf16 v8bf;
typedef __attribute__((ext_vector_type(8)))  float  v8f;

// Fixed problem shape
constexpr int B_   = 2;
constexpr int T_   = 2048;
constexpr int D_   = 1024;
constexpr int H_   = 16;
constexpr int DH_  = 64;
constexpr int M_   = B_ * T_;   // 4096 rows of combined batch

#define WMMA_BF16(a, b, c) \
  __builtin_amdgcn_wmma_f32_16x16x32_bf16(false, (a), false, (b), (short)0, (c), false, false)

static __device__ inline v16bf combine8(v8bf lo, v8bf hi) {
  v16bf r;
#pragma unroll
  for (int i = 0; i < 8; ++i) { r[i] = lo[i]; r[i + 8] = hi[i]; }
  return r;
}

// C[M,N] = A[M,K] @ W[N,K]^T  (nn.Linear, bias-free), M=4096, N=K=1024.
// Per wave: 16(M) x 128(N) tile -> 8 accumulators sharing one A fragment.
// MODE 0: fp32 row-major (M,N)
// MODE 1: bf16 per-head layout (b, h, t, dh)
// MODE 2: bf16 per-head TRANSPOSED (b, h, dh, t)
template <bool AF32, bool WF32, int MODE>
__global__ void __launch_bounds__(256)
gemm_xWt_wmma(const void* __restrict__ Ap, const void* __restrict__ Wp,
              void* __restrict__ Cp)
{
  constexpr int N = D_, K = D_;
  constexpr int TILE_N = 128, SUBS = TILE_N / 16;   // 8 accumulators
  constexpr int tiles_n = N / TILE_N;               // 8

  const int wave = (blockIdx.x * blockDim.x + threadIdx.x) >> 5;
  const int lane = threadIdx.x & 31;
  const int half = lane >> 4;
  const int l15  = lane & 15;
  const int tm = wave / tiles_n;
  const int tn = wave % tiles_n;

  const float*  Af = (const float*)Ap;
  const bf16_t* Ab = (const bf16_t*)Ap;
  const float*  Wf = (const float*)Wp;
  const bf16_t* Wb = (const bf16_t*)Wp;

  v8f acc[SUBS];
#pragma unroll
  for (int s = 0; s < SUBS; ++s)
#pragma unroll
    for (int e = 0; e < 8; ++e) acc[s][e] = 0.0f;

  const size_t arow = (size_t)(tm * 16 + l15) * K;
  size_t wrow[SUBS];
#pragma unroll
  for (int s = 0; s < SUBS; ++s) wrow[s] = (size_t)(tn * TILE_N + s * 16 + l15) * K;

  for (int kk = 0; kk < K; kk += 32) {
    // A fragment: lane = row (M); K chunks [k0, k0+8) and [k0+16, k0+24)
    const int ak0 = kk + half * 8;
    v16bf afrag;
    if constexpr (AF32) {
      if (kk + 32 < K) __builtin_prefetch(&Af[arow + kk + 32], 0, 1);
#pragma unroll
      for (int i = 0; i < 8; ++i) {
        afrag[i]     = (bf16_t)Af[arow + ak0 + i];
        afrag[i + 8] = (bf16_t)Af[arow + ak0 + 16 + i];
      }
    } else {
      if (kk + 32 < K) __builtin_prefetch(&Ab[arow + kk + 32], 0, 1);
      const bf16_t* p = Ab + arow + ak0;
      afrag = combine8(*(const v8bf*)p, *(const v8bf*)(p + 16));
    }
    // W fragments: lane = column (N); K chunks [k0, k0+8) and [k0+16, k0+24)
    const int wk0 = kk + half * 8;
#pragma unroll
    for (int s = 0; s < SUBS; ++s) {
      v16bf wfrag;
      if constexpr (WF32) {
#pragma unroll
        for (int i = 0; i < 8; ++i) { wfrag[i] = (bf16_t)Wf[wrow[s] + wk0 + i]; wfrag[i + 8] = (bf16_t)Wf[wrow[s] + wk0 + 16 + i]; }
      } else {
        const bf16_t* p = Wb + wrow[s] + wk0;
        wfrag = combine8(*(const v8bf*)p, *(const v8bf*)(p + 16));
      }
      acc[s] = WMMA_BF16(afrag, wfrag, acc[s]);
    }
  }

  // C element e of acc[s]: row = tm*16 + e + 8*half, col = tn*TILE_N + s*16 + l15
#pragma unroll
  for (int s = 0; s < SUBS; ++s) {
#pragma unroll
    for (int e = 0; e < 8; ++e) {
      const int row = tm * 16 + e + 8 * half;
      const int col = tn * TILE_N + s * 16 + l15;
      const float v = acc[s][e];
      if constexpr (MODE == 0) {
        ((float*)Cp)[(size_t)row * N + col] = v;
      } else {
        const int bi = row >> 11;          // row / T_
        const int tr = row & (T_ - 1);
        const int hd = col >> 6;           // col / DH_
        const int dc = col & (DH_ - 1);
        size_t idx;
        if constexpr (MODE == 1) idx = (((size_t)bi * H_ + hd) * T_ + tr) * DH_ + dc;
        else                     idx = (((size_t)bi * H_ + hd) * DH_ + dc) * (size_t)T_ + tr;
        ((bf16_t*)Cp)[idx] = (bf16_t)v;
      }
    }
  }
}

// Flash attention: one wave = 16 query rows of one (b,h).
// Qh, Kh: (b,h,t,64) bf16;  Vt: (b,h,64,t) bf16;  ctx out: (b,t,h*64) bf16.
__global__ void __launch_bounds__(128)
mha_flash_wmma(const bf16_t* __restrict__ Qh, const bf16_t* __restrict__ Kh,
               const bf16_t* __restrict__ Vt, const int* __restrict__ mask,
               bf16_t* __restrict__ ctx)
{
  constexpr float scale = 0.125f;      // 1/sqrt(64)
  __shared__ bf16_t pb[4][16][32];     // per-wave P staging (C-layout -> A-layout)

  const int wib  = threadIdx.x >> 5;
  const int wave = blockIdx.x * (blockDim.x >> 5) + wib;
  const int lane = threadIdx.x & 31;
  const int half = lane >> 4;
  const int l15  = lane & 15;
  constexpr int qtiles = T_ / 16;      // 128
  const int bh = wave >> 7;            // wave / qtiles
  const int qt = wave & (qtiles - 1);
  const int b = bh >> 4, h = bh & (H_ - 1);

  const bf16_t* Qp = Qh + ((size_t)bh * T_ + qt * 16) * DH_;
  const bf16_t* Kp = Kh + (size_t)bh * T_ * DH_;
  const bf16_t* Vp = Vt + (size_t)bh * DH_ * T_;

  v16bf qf[2];
#pragma unroll
  for (int f = 0; f < 2; ++f) {
    const bf16_t* p = Qp + l15 * DH_ + f * 32 + half * 8;
    qf[f] = combine8(*(const v8bf*)p, *(const v8bf*)(p + 16));
  }

  float mrow[8], lrow[8];
  v8f acc[4];
#pragma unroll
  for (int e = 0; e < 8; ++e) { mrow[e] = -1e30f; lrow[e] = 0.0f; }
#pragma unroll
  for (int c = 0; c < 4; ++c)
#pragma unroll
    for (int e = 0; e < 8; ++e) acc[c][e] = 0.0f;

  for (int kt = 0; kt < T_ / 32; ++kt) {
    const int key0 = kt * 32;
    v8f s0, s1;
#pragma unroll
    for (int e = 0; e < 8; ++e) { s0[e] = 0.0f; s1[e] = 0.0f; }

    // S = Q(16x64) @ K_tile(32x64)^T  (two 16x16 C tiles)
#pragma unroll
    for (int f = 0; f < 2; ++f) {
      const bf16_t* kp0 = Kp + (size_t)(key0 + l15) * DH_ + f * 32 + half * 8;
      v16bf kb0 = combine8(*(const v8bf*)kp0, *(const v8bf*)(kp0 + 16));
      s0 = WMMA_BF16(qf[f], kb0, s0);
      const bf16_t* kp1 = kp0 + 16 * DH_;
      v16bf kb1 = combine8(*(const v8bf*)kp1, *(const v8bf*)(kp1 + 16));
      s1 = WMMA_BF16(qf[f], kb1, s1);
    }

    // scale + mask
    const int qbase = qt * 16 + 8 * half;
#pragma unroll
    for (int e = 0; e < 8; ++e) {
      const size_t mr = (size_t)(qbase + e) * T_ + key0;
      float a0 = s0[e] * scale;
      float a1 = s1[e] * scale;
      if (mask[mr + l15] == 0)      a0 = -1e9f;
      if (mask[mr + 16 + l15] == 0) a1 = -1e9f;
      s0[e] = a0; s1[e] = a1;
    }

    // online softmax (each C row lives across the 16 lanes of a half)
#pragma unroll
    for (int e = 0; e < 8; ++e) {
      float r = fmaxf(s0[e], s1[e]);
#pragma unroll
      for (int mk = 8; mk >= 1; mk >>= 1) r = fmaxf(r, __shfl_xor(r, mk, 32));
      const float mo = mrow[e];
      const float mn = fmaxf(mo, r);
      const float co = __expf(mo - mn);
      mrow[e] = mn;
      lrow[e] *= co;
      const float p0 = __expf(s0[e] - mn);
      const float p1 = __expf(s1[e] - mn);
      s0[e] = p0; s1[e] = p1;
      float rs = p0 + p1;
#pragma unroll
      for (int mk = 8; mk >= 1; mk >>= 1) rs += __shfl_xor(rs, mk, 32);
      lrow[e] += rs;
      acc[0][e] *= co; acc[1][e] *= co; acc[2][e] *= co; acc[3][e] *= co;
    }

    // P: C-layout -> LDS -> A-layout fragment (bf16)
#pragma unroll
    for (int e = 0; e < 8; ++e) {
      const int row = e + 8 * half;
      pb[wib][row][l15]      = (bf16_t)s0[e];
      pb[wib][row][16 + l15] = (bf16_t)s1[e];
    }
    __syncthreads();
    const bf16_t* pp = &pb[wib][l15][half * 8];
    v16bf pf = combine8(*(const v8bf*)pp, *(const v8bf*)(pp + 16));

    // O += P(16x32) @ V(32x64), V pre-transposed -> contiguous B-frag loads
#pragma unroll
    for (int c = 0; c < 4; ++c) {
      const bf16_t* vp = Vp + (size_t)(c * 16 + l15) * T_ + key0 + half * 8;
      v16bf vb = combine8(*(const v8bf*)vp, *(const v8bf*)(vp + 16));
      acc[c] = WMMA_BF16(pf, vb, acc[c]);
    }
    __syncthreads();
  }

  float inv[8];
#pragma unroll
  for (int e = 0; e < 8; ++e) inv[e] = 1.0f / lrow[e];
#pragma unroll
  for (int c = 0; c < 4; ++c) {
#pragma unroll
    for (int e = 0; e < 8; ++e) {
      const int qr = qt * 16 + e + 8 * half;
      const size_t o = ((size_t)b * T_ + qr) * (size_t)D_ + h * DH_ + c * 16 + l15;
      ctx[o] = (bf16_t)(acc[c][e] * inv[e]);
    }
  }
}

extern "C" void kernel_launch(void* const* d_in, const int* in_sizes, int n_in,
                              void* d_out, int out_size, void* d_ws, size_t ws_size,
                              hipStream_t stream) {
  (void)in_sizes; (void)n_in; (void)out_size; (void)ws_size;
  const float* q  = (const float*)d_in[0];
  const float* k  = (const float*)d_in[1];
  const float* v  = (const float*)d_in[2];
  const int*   mk = (const int*)  d_in[3];
  const float* Wq = (const float*)d_in[4];
  const float* Wk = (const float*)d_in[5];
  const float* Wv = (const float*)d_in[6];
  const float* Wo = (const float*)d_in[7];

  const size_t headBytes = (size_t)B_ * H_ * T_ * DH_ * sizeof(bf16_t); // 8 MB
  char* ws = (char*)d_ws;
  bf16_t* Qh  = (bf16_t*)(ws);
  bf16_t* Kh  = (bf16_t*)(ws + headBytes);
  bf16_t* Vt  = (bf16_t*)(ws + 2 * headBytes);
  bf16_t* ctx = (bf16_t*)(ws + 3 * headBytes);   // 32 MB total workspace

  const int gemmWaves  = (M_ / 16) * (D_ / 128); // 2048 waves (16x128 tiles)
  const int gemmBlocks = gemmWaves / 8;          // 8 waves / 256-thread block
  gemm_xWt_wmma<true,  true, 1><<<gemmBlocks, 256, 0, stream>>>(q,   Wq, Qh);
  gemm_xWt_wmma<true,  true, 1><<<gemmBlocks, 256, 0, stream>>>(k,   Wk, Kh);
  gemm_xWt_wmma<true,  true, 2><<<gemmBlocks, 256, 0, stream>>>(v,   Wv, Vt);

  const int attWaves = B_ * H_ * (T_ / 16);      // 4096
  mha_flash_wmma<<<attWaves / 4, 128, 0, stream>>>(Qh, Kh, Vt, mk, ctx);

  gemm_xWt_wmma<false, true, 0><<<gemmBlocks, 256, 0, stream>>>(ctx, Wo, (float*)d_out);
}
